// CausalSelfAttention_38053410242829
// MI455X (gfx1250) — matrix operand order repaired and checked by reading, not compiled, not run on hardware
//
#include <hip/hip_runtime.h>

// CDNA5 / gfx1250 WMMA fragment types
typedef __attribute__((ext_vector_type(16))) __bf16 v16bf;
typedef __attribute__((ext_vector_type(8)))  __bf16 v8bf;
typedef __attribute__((ext_vector_type(8)))  float  v8f;
// TDM descriptor groups
typedef __attribute__((ext_vector_type(4)))  unsigned int v4u;
typedef __attribute__((ext_vector_type(8)))  int          v8i;
typedef __attribute__((ext_vector_type(4)))  int          v4i;

#define B_  2
#define T_  2048
#define C_  1024
#define H_  16
#define N3C 3072   // 3*C

// LDS row strides (elements): +8 bf16 pad = 16B, keeps b128 alignment
#define AS 40   // GEMM A tile  [m(64)][k(32)]
#define BS 40   // GEMM B tile  [n(64)][k(32)]  (transposed)
#define KS 72   // attn K tile  [key(32)][hd(64)]
#define VS 40   // attn V tile  [hd(64)][key(32)]
#define PS 40   // attn P tile  [row(16)][key(32)]

static __device__ __forceinline__ v8f wmma_bf16(v16bf a, v16bf b, v8f c) {
  return __builtin_amdgcn_wmma_f32_16x16x32_bf16(false, a, false, b, (short)0, c, false, false);
}

// A-fragment element i -> K index: lane<16 K={0..15}, lane>=16 K={16..31}
static __device__ __forceinline__ int a_kidx(int i, int half) {
  return half * 16 + i;
}

#if __has_builtin(__builtin_amdgcn_tensor_load_to_lds)
#define USE_TDM 1
#else
#define USE_TDM 0
#endif

#if USE_TDM
// Issue one 2D TDM tile load: global (strided rows) -> LDS (rows padded).
// pad_ivl: rows of 2^(pad_ivl+1) DWORDs between pads; pad_amt: (pad_amt+1) DWORDs pad.
// This toolchain exposes the 6-arg builtin: (g0 v4u, g1 v8i, g2 v4i, g3 v4i, v8i, cpol)
static __device__ __forceinline__ void tdm_load_2d(
    const __bf16* gptr, void* lds_ptr,
    unsigned tensor_d0, unsigned tensor_d1,
    unsigned tile_d0, unsigned tile_d1,
    unsigned stride0_elems, unsigned pad_ivl, unsigned pad_amt) {
  unsigned long long ga = (unsigned long long)(uintptr_t)gptr;
  unsigned lds_off = (unsigned)(uintptr_t)lds_ptr;  // flat shared addr[31:0] = LDS byte offset
  v4u g0;
  g0[0] = 1u;                                       // count=1, user descriptor
  g0[1] = lds_off;
  g0[2] = (unsigned)ga;                             // global_addr[31:0]  (bits 95:64)
  g0[3] = (unsigned)(ga >> 32) | (2u << 30);        // global_addr[56:32] | type=2
  v8i g1;
  g1[0] = (int)((1u << 16) |                        // data_size = 2 bytes
                (1u << 20) |                        // pad_enable
                (pad_ivl << 22) | (pad_amt << 25));
  g1[1] = (int)((tensor_d0 & 0xFFFFu) << 16);       // tensor_dim0[15:0]
  g1[2] = (int)((tensor_d0 >> 16) | ((tensor_d1 & 0xFFFFu) << 16));
  g1[3] = (int)((tensor_d1 >> 16) | (tile_d0 << 16));
  g1[4] = (int)tile_d1;                             // tile_dim1 (tile_dim2 = 0)
  g1[5] = (int)stride0_elems;                       // tensor_dim0_stride lo32
  g1[6] = 0;                                        // stride0 hi16 | stride1 lo16
  g1[7] = 0;
  v4i gz4 = {0, 0, 0, 0};
  v8i gz8 = {0, 0, 0, 0, 0, 0, 0, 0};
  __builtin_amdgcn_tensor_load_to_lds(g0, g1, gz4, gz4, gz8, 0);
}
#endif

// ---------------------------------------------------------------------------
// Kernel 1: KQV = x @ w_kqv + b_kqv -> Q [B,H,T,64], K [B,H,T,64], Vt [B,H,64,T]
// ---------------------------------------------------------------------------
__global__ __launch_bounds__(256) void kqv_gemm_kernel(
    const float* __restrict__ x, const float* __restrict__ w,
    const float* __restrict__ bias,
    __bf16* __restrict__ Qo, __bf16* __restrict__ Ko, __bf16* __restrict__ Vto) {
  const int n0 = blockIdx.x * 64;   // N = 3072
  const int m0 = blockIdx.y * 64;   // M = 4096
  const int tid  = threadIdx.x;
  const int wave = tid >> 5;
  const int lane = tid & 31;
  const int half = lane >> 4;
  const int lcol = lane & 15;
  const int wm = wave >> 1;
  const int wn = wave & 1;

  __shared__ __bf16 a_lds[2][64 * AS];   // [m][k]
  __shared__ __bf16 b_lds[2][64 * BS];   // [n][k] (transposed)

  auto load_tiles = [&](int kk, int buf) {
    {  // A tile: 64x32 f32 -> bf16, row-major, vectorized
      int r = tid >> 2, c = (tid & 3) * 8;
      const float4* src = reinterpret_cast<const float4*>(&x[(size_t)(m0 + r) * C_ + kk + c]);
      float4 f0 = src[0], f1 = src[1];
      v8bf pk;
      pk[0] = (__bf16)f0.x; pk[1] = (__bf16)f0.y; pk[2] = (__bf16)f0.z; pk[3] = (__bf16)f0.w;
      pk[4] = (__bf16)f1.x; pk[5] = (__bf16)f1.y; pk[6] = (__bf16)f1.z; pk[7] = (__bf16)f1.w;
      *reinterpret_cast<v8bf*>(&a_lds[buf][r * AS + c]) = pk;
    }
    {  // B tile: 32x64 f32, stored transposed [n][k]
      int r = tid >> 3, c = (tid & 7) * 8;
      const float4* src = reinterpret_cast<const float4*>(&w[(size_t)(kk + r) * N3C + n0 + c]);
      float4 f0 = src[0], f1 = src[1];
      float fv[8] = {f0.x, f0.y, f0.z, f0.w, f1.x, f1.y, f1.z, f1.w};
#pragma unroll
      for (int j = 0; j < 8; ++j) b_lds[buf][(c + j) * BS + r] = (__bf16)fv[j];
    }
  };

  v8f acc[2];
#pragma unroll
  for (int h = 0; h < 2; ++h)
#pragma unroll
    for (int r = 0; r < 8; ++r) acc[h][r] = 0.0f;

  load_tiles(0, 0);
  const int NIT = C_ / 32;
  for (int it = 0; it < NIT; ++it) {
    __syncthreads();
    if (it + 1 < NIT) load_tiles((it + 1) * 32, (it + 1) & 1);
    const int buf = it & 1;

    v16bf a;
#pragma unroll
    for (int i = 0; i < 16; ++i)
      a[i] = a_lds[buf][(wm * 16 + lcol) * AS + a_kidx(i, half)];
#pragma unroll
    for (int nh = 0; nh < 2; ++nh) {
      v16bf b;
#pragma unroll
      for (int e = 0; e < 16; ++e)
        b[e] = b_lds[buf][(wn * 32 + nh * 16 + lcol) * BS + half * 16 + e];
      acc[nh] = wmma_bf16(a, b, acc[nh]);
    }
  }

  // Epilogue: +bias, per-head 192-chunk routing (k | q | v)
#pragma unroll
  for (int nh = 0; nh < 2; ++nh) {
#pragma unroll
    for (int r = 0; r < 8; ++r) {
      int n = n0 + wn * 32 + nh * 16 + lcol;
      int m = m0 + wm * 16 + half * 8 + r;
      float v = acc[nh][r] + bias[n];
      int h  = n / 192;
      int rr = n % 192;
      int b  = m >> 11;
      int t  = m & (T_ - 1);
      size_t bh = (size_t)(b * H_ + h);
      __bf16 bv = (__bf16)v;
      if (rr < 64)        Ko [(bh * T_ + t) * 64 + rr]          = bv;  // K natural
      else if (rr < 128)  Qo [(bh * T_ + t) * 64 + (rr - 64)]   = bv;  // Q
      else                Vto[(bh * 64 + (rr - 128)) * T_ + t]  = bv;  // V transposed
    }
  }
}

// ---------------------------------------------------------------------------
// Kernel 2: causal flash attention. grid = (B*H, T/128), 8 waves/block.
//   K/V tiles staged via Tensor Data Mover (gfx1250 TDM) when available.
// ---------------------------------------------------------------------------
__global__ __launch_bounds__(256) void flash_attn_kernel(
    const __bf16* __restrict__ Q, const __bf16* __restrict__ K,
    const __bf16* __restrict__ Vt, __bf16* __restrict__ Yout) {
  const int bh   = blockIdx.x;
  const int qblk = blockIdx.y;
  const int tid  = threadIdx.x;
  const int wave = tid >> 5;
  const int lane = tid & 31;
  const int half = lane >> 4;
  const int lcol = lane & 15;
  const int q0   = qblk * 128 + wave * 16;

  __shared__ __bf16 ktile[2][32 * KS];   // [key][hd], rows padded to 144B
  __shared__ __bf16 vtile[2][64 * VS];   // [hd][key], rows padded to 80B
  __shared__ __bf16 ptile[8][16 * PS];   // per-wave P transpose staging

  const __bf16* Qb  = Q  + (size_t)bh * T_ * 64;
  const __bf16* Kb  = K  + (size_t)bh * T_ * 64;
  const __bf16* Vtb = Vt + (size_t)bh * 64 * T_;

#if USE_TDM
  // K tile rows: 64 elems = 32 DWORDs (pad_ivl=4), pad 8 elems = 4 DWORDs (pad_amt=3)
  // V tile rows: 32 elems = 16 DWORDs (pad_ivl=3), pad 8 elems = 4 DWORDs (pad_amt=3)
  auto issue_tiles = [&](int kt, int buf) {
    tdm_load_2d(Kb + (size_t)(kt * 32) * 64, &ktile[buf][0],
                64u, (unsigned)T_, 64u, 32u, 64u, 4u, 3u);
    tdm_load_2d(Vtb + (size_t)(kt * 32), &vtile[buf][0],
                (unsigned)T_, 64u, 32u, 64u, (unsigned)T_, 3u, 3u);
  };
#else
  auto load_tiles = [&](int kt, int buf) {
    {  // K tile: 32(key) x 64(hd)
      int r = tid >> 3, c = (tid & 7) * 8;
      *reinterpret_cast<v8bf*>(&ktile[buf][r * KS + c]) =
          *reinterpret_cast<const v8bf*>(&Kb[(size_t)(kt * 32 + r) * 64 + c]);
    }
    {  // V tile: 64(hd) x 32(key)
      int r = tid >> 2, c = (tid & 3) * 8;
      *reinterpret_cast<v8bf*>(&vtile[buf][r * VS + c]) =
          *reinterpret_cast<const v8bf*>(&Vtb[(size_t)r * T_ + kt * 32 + c]);
    }
  };
#endif

  // Q A-fragments (hd -> two K-chunks of 32); contiguous 16-element pieces
  v16bf aq[2];
  {
    int qrow = q0 + lcol;
#pragma unroll
    for (int c = 0; c < 2; ++c)
#pragma unroll
      for (int i = 0; i < 16; ++i)
        aq[c][i] = Qb[(size_t)qrow * 64 + c * 32 + a_kidx(i, half)];
  }

  v8f o[4];
#pragma unroll
  for (int t = 0; t < 4; ++t)
#pragma unroll
    for (int r = 0; r < 8; ++r) o[t][r] = 0.0f;
  float m_i[8], l_i[8];
#pragma unroll
  for (int r = 0; r < 8; ++r) { m_i[r] = -1e30f; l_i[r] = 0.0f; }

  const int nkt = qblk * 4 + 4;

#if USE_TDM
  if (wave == 0) issue_tiles(0, 0);
#else
  load_tiles(0, 0);
#endif

  for (int kt = 0; kt < nkt; ++kt) {
    const int buf = kt & 1;
#if USE_TDM
    __syncthreads();                 // all waves done with buf^1 before overwrite
    if (wave == 0) {
      if (kt + 1 < nkt) {
        issue_tiles(kt + 1, buf ^ 1);
        __builtin_amdgcn_s_wait_tensorcnt(2);  // tile kt complete (in-order TDM)
      } else {
        __builtin_amdgcn_s_wait_tensorcnt(0);
      }
    }
    __syncthreads();                 // publish tile kt to all waves
#else
    __syncthreads();
    if (kt + 1 < nkt) load_tiles(kt + 1, buf ^ 1);
#endif

    if (kt * 32 <= q0 + 15) {        // wave-uniform causal skip
      // S = Q * K^T : B[k=hd][n=key] = ktile[key][hd] -> contiguous gathers
      v8f s[2];
#pragma unroll
      for (int h = 0; h < 2; ++h) {
#pragma unroll
        for (int r = 0; r < 8; ++r) s[h][r] = 0.0f;
#pragma unroll
        for (int c = 0; c < 2; ++c) {
          v16bf bk;
#pragma unroll
          for (int e = 0; e < 16; ++e)
            bk[e] = ktile[buf][(h * 16 + lcol) * KS + c * 32 + half * 16 + e];
          s[h] = wmma_bf16(aq[c], bk, s[h]);
        }
      }
      // scale + causal mask
      const float scale = 0.125f;
#pragma unroll
      for (int h = 0; h < 2; ++h)
#pragma unroll
        for (int r = 0; r < 8; ++r) {
          int row = q0 + half * 8 + r;
          int col = kt * 32 + h * 16 + lcol;
          float v = s[h][r] * scale;
          s[h][r] = (col > row) ? -1e30f : v;
        }
      // online softmax (row stats across the 16 lanes of each half)
      float tmax[8], tsum[8];
#pragma unroll
      for (int r = 0; r < 8; ++r) tmax[r] = fmaxf(s[0][r], s[1][r]);
#pragma unroll
      for (int off = 1; off < 16; off <<= 1)
#pragma unroll
        for (int r = 0; r < 8; ++r)
          tmax[r] = fmaxf(tmax[r], __shfl_xor(tmax[r], off, 16));
      float alpha[8];
#pragma unroll
      for (int r = 0; r < 8; ++r) {
        float mn = fmaxf(m_i[r], tmax[r]);
        alpha[r] = __expf(m_i[r] - mn);
        m_i[r] = mn;
      }
#pragma unroll
      for (int h = 0; h < 2; ++h)
#pragma unroll
        for (int r = 0; r < 8; ++r) s[h][r] = __expf(s[h][r] - m_i[r]);
#pragma unroll
      for (int r = 0; r < 8; ++r) tsum[r] = s[0][r] + s[1][r];
#pragma unroll
      for (int off = 1; off < 16; off <<= 1)
#pragma unroll
        for (int r = 0; r < 8; ++r) tsum[r] += __shfl_xor(tsum[r], off, 16);
#pragma unroll
      for (int r = 0; r < 8; ++r) l_i[r] = l_i[r] * alpha[r] + tsum[r];
#pragma unroll
      for (int t = 0; t < 4; ++t)
#pragma unroll
        for (int r = 0; r < 8; ++r) o[t][r] *= alpha[r];

      // P: C-frag -> A-frag via per-wave LDS (same-wave DS ops are in-order)
      __bf16* pw = &ptile[wave][0];
#pragma unroll
      for (int r = 0; r < 8; ++r) {
        pw[(half * 8 + r) * PS + lcol]      = (__bf16)s[0][r];
        pw[(half * 8 + r) * PS + 16 + lcol] = (__bf16)s[1][r];
      }
      v16bf pa;
#pragma unroll
      for (int i = 0; i < 16; ++i)
        pa[i] = pw[lcol * PS + a_kidx(i, half)];

      // O += P * V : B[k=key][n=hd] = vtile[hd][key] -> contiguous gathers
#pragma unroll
      for (int t = 0; t < 4; ++t) {
        v16bf bv;
#pragma unroll
        for (int e = 0; e < 16; ++e)
          bv[e] = vtile[buf][(t * 16 + lcol) * VS + half * 16 + e];
        o[t] = wmma_bf16(pa, bv, o[t]);
      }
    }
  }

  // Epilogue: normalize, write y bf16 [B*T, C] (c = h*64 + d)
  {
    int b = bh >> 4, h = bh & 15;
    float linv[8];
#pragma unroll
    for (int r = 0; r < 8; ++r) linv[r] = 1.0f / l_i[r];
#pragma unroll
    for (int t = 0; t < 4; ++t)
#pragma unroll
      for (int r = 0; r < 8; ++r) {
        int row = q0 + half * 8 + r;
        Yout[((size_t)(b * T_ + row)) * C_ + h * 64 + t * 16 + lcol] =
            (__bf16)(o[t][r] * linv[r]);
      }
  }
}

// ---------------------------------------------------------------------------
// Kernel 3: out = y @ w_proj + b_proj   (y bf16 [4096,1024], out f32)
// ---------------------------------------------------------------------------
__global__ __launch_bounds__(256) void proj_gemm_kernel(
    const __bf16* __restrict__ y, const float* __restrict__ w,
    const float* __restrict__ bias, float* __restrict__ out) {
  const int n0 = blockIdx.x * 64;
  const int m0 = blockIdx.y * 64;
  const int tid  = threadIdx.x;
  const int wave = tid >> 5;
  const int lane = tid & 31;
  const int half = lane >> 4;
  const int lcol = lane & 15;
  const int wm = wave >> 1;
  const int wn = wave & 1;

  __shared__ __bf16 a_lds[2][64 * AS];
  __shared__ __bf16 b_lds[2][64 * BS];

  auto load_tiles = [&](int kk, int buf) {
    {  // A tile: bf16 y, straight b128 copy
      int r = tid >> 2, c = (tid & 3) * 8;
      *reinterpret_cast<v8bf*>(&a_lds[buf][r * AS + c]) =
          *reinterpret_cast<const v8bf*>(&y[(size_t)(m0 + r) * C_ + kk + c]);
    }
    {  // B tile: f32 w, stored transposed [n][k]
      int r = tid >> 3, c = (tid & 7) * 8;
      const float4* src = reinterpret_cast<const float4*>(&w[(size_t)(kk + r) * C_ + n0 + c]);
      float4 f0 = src[0], f1 = src[1];
      float fv[8] = {f0.x, f0.y, f0.z, f0.w, f1.x, f1.y, f1.z, f1.w};
#pragma unroll
      for (int j = 0; j < 8; ++j) b_lds[buf][(c + j) * BS + r] = (__bf16)fv[j];
    }
  };

  v8f acc[2];
#pragma unroll
  for (int h = 0; h < 2; ++h)
#pragma unroll
    for (int r = 0; r < 8; ++r) acc[h][r] = 0.0f;

  load_tiles(0, 0);
  const int NIT = C_ / 32;
  for (int it = 0; it < NIT; ++it) {
    __syncthreads();
    if (it + 1 < NIT) load_tiles((it + 1) * 32, (it + 1) & 1);
    const int buf = it & 1;

    v16bf a;
#pragma unroll
    for (int i = 0; i < 16; ++i)
      a[i] = a_lds[buf][(wm * 16 + lcol) * AS + a_kidx(i, half)];
#pragma unroll
    for (int nh = 0; nh < 2; ++nh) {
      v16bf b;
#pragma unroll
      for (int e = 0; e < 16; ++e)
        b[e] = b_lds[buf][(wn * 32 + nh * 16 + lcol) * BS + half * 16 + e];
      acc[nh] = wmma_bf16(a, b, acc[nh]);
    }
  }

#pragma unroll
  for (int nh = 0; nh < 2; ++nh)
#pragma unroll
    for (int r = 0; r < 8; ++r) {
      int n = n0 + wn * 32 + nh * 16 + lcol;
      int m = m0 + wm * 16 + half * 8 + r;
      out[(size_t)m * C_ + n] = acc[nh][r] + bias[n];
    }
}

// ---------------------------------------------------------------------------
extern "C" void kernel_launch(void* const* d_in, const int* in_sizes, int n_in,
                              void* d_out, int out_size, void* d_ws, size_t ws_size,
                              hipStream_t stream) {
  const float* x      = (const float*)d_in[0];
  // d_in[1] = att_mask (causal tril) -- implemented analytically, unused
  const float* w_kqv  = (const float*)d_in[2];
  const float* b_kqv  = (const float*)d_in[3];
  const float* w_proj = (const float*)d_in[4];
  const float* b_proj = (const float*)d_in[5];
  float* out = (float*)d_out;

  char* ws = (char*)d_ws;
  const size_t SEG = (size_t)B_ * H_ * T_ * 64 * sizeof(__bf16);  // 8 MB
  __bf16* Q  = (__bf16*)(ws);
  __bf16* K  = (__bf16*)(ws + SEG);
  __bf16* Vt = (__bf16*)(ws + 2 * SEG);
  __bf16* Y  = (__bf16*)(ws + 3 * SEG);

  dim3 blk(256, 1, 1);
  dim3 g1(N3C / 64, (B_ * T_) / 64, 1);
  hipLaunchKernelGGL(kqv_gemm_kernel, g1, blk, 0, stream, x, w_kqv, b_kqv, Q, K, Vt);

  dim3 g2(B_ * H_, T_ / 128, 1);
  hipLaunchKernelGGL(flash_attn_kernel, g2, blk, 0, stream, Q, K, Vt, Y);

  dim3 g3(C_ / 64, (B_ * T_) / 64, 1);
  hipLaunchKernelGGL(proj_gemm_kernel, g3, blk, 0, stream, Y, w_proj, b_proj, out);
}
